// VQ_EMA_layer_1099511627869
// MI455X (gfx1250) — matrix operand order repaired and checked by reading, not compiled, not run on hardware
//
#include <hip/hip_runtime.h>
#include <hip/hip_bf16.h>

typedef __attribute__((ext_vector_type(16))) _Float16 v16h;
typedef __attribute__((ext_vector_type(8)))  _Float16 v8h;
typedef __attribute__((ext_vector_type(8)))  float    v8f;
typedef __attribute__((ext_vector_type(4)))  int      v4i;

#define VQ_B 16
#define VQ_T 1024
#define VQ_D 256
#define VQ_K 1024
#define VQ_N (VQ_B * VQ_T)          // 16384 rows
#define DECAY 0.99f
#define VQ_EPS 1e-5f
#define WAVES_PER_WG 8              // 8 waves -> 128 rows per WG

// ---------------- d_out layout (tuple order, flat) ----------------
// quantized [N*D], loss [1], embed_new [K*D], cs [K], ema_w_new [K*D]
#define OUT_Q_OFF     0
#define OUT_LOSS_OFF  (VQ_N * VQ_D)                 // 4194304
#define OUT_EMBED_OFF (OUT_LOSS_OFF + 1)            // 4194305
#define OUT_CS_OFF    (OUT_EMBED_OFF + VQ_K * VQ_D) // 4456449
#define OUT_EMAW_OFF  (OUT_CS_OFF + VQ_K)           // 4457473

// ---------------- workspace layout (bytes) ----------------
#define WS_XH_OFF     0u                                     // f16 [N*D]  8 MB
#define WS_WH_OFF     (WS_XH_OFF + VQ_N * VQ_D * 2u)         // f16 [K*D]  512 KB
#define WS_HW2_OFF    (WS_WH_OFF + VQ_K * VQ_D * 2u)         // f32 [K]
#define WS_IDX_OFF    (WS_HW2_OFF + VQ_K * 4u)               // u32 [N]
#define WS_ZERO_OFF   (WS_IDX_OFF + VQ_N * 4u)               // start of zeroed region
#define WS_COUNTS_OFF (WS_ZERO_OFF)                          // f32 [K]
#define WS_DW_OFF     (WS_COUNTS_OFF + VQ_K * 4u)            // f32 [K*D]
#define WS_LOSS_OFF   (WS_DW_OFF + VQ_K * VQ_D * 4u)         // f32 [1]
#define WS_CS_OFF     (WS_LOSS_OFF + 4u)                     // f32 [K]
#define ZERO_FLOATS   (VQ_K + VQ_K * VQ_D + 1)               // counts + dw + lossAcc

#if __has_builtin(__builtin_amdgcn_global_load_async_to_lds_b128)
#define HAS_ASYNC_LDS 1
// async copy 16 bytes global -> LDS, tracked by ASYNCcnt (CDNA5).
// Probe-discovered param types: (AS1 int4*, AS3 ptr, int offset, int cpol).
#define ASYNC_CP16(gptr, lptr)                                              \
    __builtin_amdgcn_global_load_async_to_lds_b128(                         \
        (__attribute__((address_space(1))) v4i*)(gptr),                     \
        (__attribute__((address_space(3))) v4i*)(lptr), 0, 0)
#else
#define HAS_ASYNC_LDS 0
#endif

// ================= kernel 0: zero accumulators =================
__global__ void kZero(float* __restrict__ p, int n) {
    int i = blockIdx.x * blockDim.x + threadIdx.x;
    if (i < n) p[i] = 0.0f;
}

// ================= kernel 1: x f32 -> f16 =================
__global__ void kCvtX(const float* __restrict__ x, _Float16* __restrict__ xh, int n) {
    int i = blockIdx.x * blockDim.x + threadIdx.x;
    if (i < n) xh[i] = (_Float16)x[i];
}

// ================= kernel 2: codebook f32->f16 + 0.5*||w||^2 =================
__global__ __launch_bounds__(256) void kPrepW(const float* __restrict__ W,
                                              _Float16* __restrict__ Wh,
                                              float* __restrict__ halfW2) {
    __shared__ float red[8];
    const int k = blockIdx.x;
    const int d = threadIdx.x;
    float w = W[(size_t)k * VQ_D + d];
    Wh[(size_t)k * VQ_D + d] = (_Float16)w;
    float p = w * w;
#pragma unroll
    for (int m = 16; m >= 1; m >>= 1) p += __shfl_xor(p, m, 32);
    if ((d & 31) == 0) red[d >> 5] = p;
    __syncthreads();
    if (d == 0) {
        float s = 0.0f;
#pragma unroll
        for (int i = 0; i < 8; ++i) s += red[i];
        halfW2[k] = 0.5f * s;
    }
}

// ================= kernel 3: WMMA distance GEMM + argmin =================
// Wave handles 16 rows; WG (8 waves) handles 128 rows; 64 column tiles of 16 codes.
// dist = ||x||^2 - 2*(x.w - 0.5||w||^2)  ->  argmin dist == argmax (x.w - 0.5||w||^2)
// Double-buffered LDS staging (async global->LDS when available), one barrier/tile,
// and two independent WMMA accumulator chains to break the RAW dependency chain.
__global__ __launch_bounds__(256) void kDistArgmin(const _Float16* __restrict__ Xh,
                                                   const _Float16* __restrict__ Wh,
                                                   const float* __restrict__ halfW2,
                                                   unsigned* __restrict__ idxOut) {
    // B tile: 16 cols x 256 d, padded row stride 272 halves (544 B) to spread banks
    __shared__ __align__(32) _Float16 sB[2][16 * 272];
    __shared__ float sW2[2][16];

    const int tid  = threadIdx.x;
    const int wave = tid >> 5;
    const int lane = tid & 31;
    const int l16  = lane & 15;
    const int half = lane >> 4;
    const int rowBase = (blockIdx.x * WAVES_PER_WG + wave) * 16;

    // staging coordinates: thread t -> col t/16, 16-half chunk t%16 (32 B per thread)
    const int scol = tid >> 4, schunk = tid & 15;
    const _Float16* gW   = Wh + (size_t)scol * VQ_D + schunk * 16;  // + colBase*VQ_D per tile
    const int       loff = scol * 272 + schunk * 16;

    // ---- preload A panel (16 rows x 256 d, f16) into registers: 8 x v16h ----
    // A layout (matches B): lane holds M=lane%16; lanes<16: K=s*32+{0..15};
    //                      lanes>=16: K=s*32+{16..31}
    v16h A[8];
    {
        const _Float16* xr = Xh + (size_t)(rowBase + l16) * VQ_D;
        const int aoff = half * 16;
#pragma unroll
        for (int s = 0; s < 8; ++s) {
            v8h lo = *(const v8h*)(xr + s * 32 + aoff);
            v8h hi = *(const v8h*)(xr + s * 32 + aoff + 8);
            v16h a;
#pragma unroll
            for (int i = 0; i < 8; ++i) { a[i] = lo[i]; a[8 + i] = hi[i]; }
            A[s] = a;
        }
    }

    // ---- prologue: stage tile 0 into buffer 0 ----
#if HAS_ASYNC_LDS
    ASYNC_CP16(gW,     &sB[0][loff]);
    ASYNC_CP16(gW + 8, &sB[0][loff + 8]);
#else
    {
        const v8h* src = (const v8h*)gW;
        v8h w0 = src[0], w1 = src[1];
        *(v8h*)(&sB[0][loff])     = w0;
        *(v8h*)(&sB[0][loff + 8]) = w1;
    }
#endif
    if (tid < 16) sW2[0][tid] = halfW2[tid];
#if HAS_ASYNC_LDS
    asm volatile("s_wait_asynccnt 0x0" ::: "memory");
#endif
    __syncthreads();

    float    bestVal[8];
    unsigned bestIdx[8];
#pragma unroll
    for (int r = 0; r < 8; ++r) { bestVal[r] = -3.0e38f; bestIdx[r] = 0u; }

    for (int ct = 0; ct < 64; ++ct) {
        const int buf = ct & 1;
        const int colBase = ct * 16;

        // ---- issue staging of tile ct+1 into the other buffer (overlaps compute) ----
        float pw2 = 0.0f;
#if HAS_ASYNC_LDS
        if (ct < 63) {
            const _Float16* g = gW + (size_t)(colBase + 16) * VQ_D;
            ASYNC_CP16(g,     &sB[buf ^ 1][loff]);
            ASYNC_CP16(g + 8, &sB[buf ^ 1][loff + 8]);
        }
#else
        v8h p0 = {}, p1 = {};
        if (ct < 63) {
            const v8h* src = (const v8h*)(gW + (size_t)(colBase + 16) * VQ_D);
            p0 = src[0];
            p1 = src[1];
        }
#endif
        if (ct < 63 && tid < 16) pw2 = halfW2[colBase + 16 + tid];

        // ---- compute: two independent WMMA chains over D=256 ----
        v8f c0 = {}, c1 = {};
#pragma unroll
        for (int s = 0; s < 4; ++s) {
            // B layout: lane holds N=lane%16; lanes<16: K=0..15, lanes>=16: K=16..31
            v16h b0 = *(const v16h*)(&sB[buf][l16 * 272 + (2 * s) * 32 + half * 16]);
            v16h b1 = *(const v16h*)(&sB[buf][l16 * 272 + (2 * s + 1) * 32 + half * 16]);
            c0 = __builtin_amdgcn_wmma_f32_16x16x32_f16(
                     false, A[2 * s],     false, b0, (short)0, c0, false, false);
            c1 = __builtin_amdgcn_wmma_f32_16x16x32_f16(
                     false, A[2 * s + 1], false, b1, (short)0, c1, false, false);
        }

        const float    hw2 = sW2[buf][l16];
        const unsigned col = (unsigned)(colBase + l16);
#pragma unroll
        for (int r = 0; r < 8; ++r) {
            float t = (c0[r] + c1[r]) - hw2;            // score; argmax == argmin dist
            if (t > bestVal[r]) { bestVal[r] = t; bestIdx[r] = col; }
        }

        // ---- finish staging + single barrier (double buffer needs only one) ----
        if (ct < 63) {
#if !HAS_ASYNC_LDS
            *(v8h*)(&sB[buf ^ 1][loff])     = p0;
            *(v8h*)(&sB[buf ^ 1][loff + 8]) = p1;
#endif
            if (tid < 16) sW2[buf ^ 1][tid] = pw2;
#if HAS_ASYNC_LDS
            asm volatile("s_wait_asynccnt 0x0" ::: "memory");
#endif
            __syncthreads();
        }
    }

    // cross-lane argmax within each 16-lane half (C layout: lanes 0-15 row r, 16-31 row r+8)
#pragma unroll
    for (int r = 0; r < 8; ++r) {
        float v = bestVal[r];
        unsigned bi = bestIdx[r];
#pragma unroll
        for (int m = 1; m < 16; m <<= 1) {
            float    ov = __shfl_xor(v, m, 32);
            unsigned oi = (unsigned)__shfl_xor((int)bi, m, 32);
            if (ov > v || (ov == v && oi < bi)) { v = ov; bi = oi; }  // first-min tiebreak
        }
        if (l16 == 0) idxOut[rowBase + half * 8 + r] = bi;
    }
}

// ================= kernel 4: gather + loss + counts + dw scatter =================
__global__ __launch_bounds__(256) void kGatherLoss(const float* __restrict__ x,
                                                   const float* __restrict__ W,
                                                   const unsigned* __restrict__ idx,
                                                   float* __restrict__ qOut,
                                                   float* __restrict__ counts,
                                                   float* __restrict__ dw,
                                                   float* __restrict__ lossAcc) {
    const int n = blockIdx.x;
    const int d = threadIdx.x;
    const unsigned k = idx[n];
    const float xv = x[(size_t)n * VQ_D + d];
    const float qv = W[(size_t)k * VQ_D + d];
    qOut[(size_t)n * VQ_D + d] = qv;
    atomicAdd(&dw[(size_t)k * VQ_D + d], xv);
    float diff = xv - qv;
    float p = diff * diff;
#pragma unroll
    for (int m = 16; m >= 1; m >>= 1) p += __shfl_xor(p, m, 32);
    if ((threadIdx.x & 31) == 0) atomicAdd(lossAcc, p);
    if (threadIdx.x == 0) atomicAdd(&counts[k], 1.0f);
}

// ================= kernel 5: cluster-size EMA + smoothing + loss finalize =================
__global__ __launch_bounds__(1024) void kCs(const float* __restrict__ emaCs,
                                            const float* __restrict__ counts,
                                            const float* __restrict__ lossAcc,
                                            float* __restrict__ csOut,
                                            float* __restrict__ csWs,
                                            float* __restrict__ lossOut) {
    __shared__ float red[32];
    const int k = threadIdx.x;
    float cs = emaCs[k] * DECAY + (1.0f - DECAY) * counts[k];
    float s = cs;
#pragma unroll
    for (int m = 16; m >= 1; m >>= 1) s += __shfl_xor(s, m, 32);
    if ((k & 31) == 0) red[k >> 5] = s;
    __syncthreads();
    if (k < 32) {
        float t = red[k];
#pragma unroll
        for (int m = 16; m >= 1; m >>= 1) t += __shfl_xor(t, m, 32);
        if (k == 0) red[0] = t;
    }
    __syncthreads();
    const float n = red[0];
    const float csf = (cs + VQ_EPS) / (n + (float)VQ_K * VQ_EPS) * n;
    csOut[k] = csf;
    csWs[k]  = csf;
    if (k == 0) lossOut[0] = 0.25f * lossAcc[0] / (float)((size_t)VQ_N * VQ_D);
}

// ================= kernel 6: ema_w update + embed normalize =================
__global__ __launch_bounds__(256) void kEma(const float* __restrict__ emaW,
                                            const float* __restrict__ dw,
                                            const float* __restrict__ csWs,
                                            float* __restrict__ emaOut,
                                            float* __restrict__ embedOut) {
    const int k = blockIdx.x;
    const int d = threadIdx.x;
    const size_t i = (size_t)k * VQ_D + d;
    const float e = emaW[i] * DECAY + (1.0f - DECAY) * dw[i];
    emaOut[i]   = e;
    embedOut[i] = e / csWs[k];
}

extern "C" void kernel_launch(void* const* d_in, const int* in_sizes, int n_in,
                              void* d_out, int out_size, void* d_ws, size_t ws_size,
                              hipStream_t stream) {
    const float* x      = (const float*)d_in[0];   // [B,T,D]
    const float* W      = (const float*)d_in[1];   // [K,D]
    const float* emaCs  = (const float*)d_in[2];   // [K]
    const float* emaW   = (const float*)d_in[3];   // [K,D]

    char* ws = (char*)d_ws;
    _Float16* Xh      = (_Float16*)(ws + WS_XH_OFF);
    _Float16* Wh      = (_Float16*)(ws + WS_WH_OFF);
    float*    halfW2  = (float*)(ws + WS_HW2_OFF);
    unsigned* idx     = (unsigned*)(ws + WS_IDX_OFF);
    float*    counts  = (float*)(ws + WS_COUNTS_OFF);
    float*    dw      = (float*)(ws + WS_DW_OFF);
    float*    lossAcc = (float*)(ws + WS_LOSS_OFF);
    float*    csWs    = (float*)(ws + WS_CS_OFF);

    float* out       = (float*)d_out;
    float* qOut      = out + OUT_Q_OFF;
    float* lossOut   = out + OUT_LOSS_OFF;
    float* embedOut  = out + OUT_EMBED_OFF;
    float* csOut     = out + OUT_CS_OFF;
    float* emaOut    = out + OUT_EMAW_OFF;

    // 0) zero accumulators (counts + dw + lossAcc are contiguous)
    kZero<<<(ZERO_FLOATS + 255) / 256, 256, 0, stream>>>(counts, ZERO_FLOATS);
    // 1) x -> f16
    kCvtX<<<(VQ_N * VQ_D + 255) / 256, 256, 0, stream>>>(x, Xh, VQ_N * VQ_D);
    // 2) codebook -> f16 + 0.5||w||^2
    kPrepW<<<VQ_K, 256, 0, stream>>>(W, Wh, halfW2);
    // 3) WMMA distance GEMM + argmin
    kDistArgmin<<<VQ_N / (16 * WAVES_PER_WG), 256, 0, stream>>>(Xh, Wh, halfW2, idx);
    // 4) gather quantized + loss partials + counts + dw scatter
    kGatherLoss<<<VQ_N, VQ_D, 0, stream>>>(x, W, idx, qOut, counts, dw, lossAcc);
    // 5) cluster-size EMA + smoothing + loss finalize
    kCs<<<1, VQ_K, 0, stream>>>(emaCs, counts, lossAcc, csOut, csWs, lossOut);
    // 6) ema_w update + normalized codebook
    kEma<<<VQ_K, VQ_D, 0, stream>>>(emaW, dw, csWs, emaOut, embedOut);
}
